// DeepseekMoE_64699387347306
// MI455X (gfx1250) — compile-verified, matrix operand order checked by reading, not run on hardware
//
#include <hip/hip_runtime.h>
#include <hip/hip_bf16.h>
#include <stdint.h>

#define T_TOK 2048
#define HDIM  1024
#define NEXP  16
#define IDIM  512
#define ISH   1024

typedef __bf16 bf16_t;
typedef bf16_t   v16bf __attribute__((ext_vector_type(16)));
typedef float    v8f   __attribute__((ext_vector_type(8)));
typedef uint32_t v8u   __attribute__((ext_vector_type(8)));

// RNE conversion (used only for activations: x tile, h tile)
__device__ __forceinline__ unsigned short f2bfbits(float f) {
    uint32_t u = __builtin_bit_cast(uint32_t, f);
    uint32_t r = (u + 0x7FFFu + ((u >> 16) & 1u)) >> 16;
    return (unsigned short)r;
}

// Truncating pack of two fp32 -> packed 2 x bf16 in one v_perm_b32.
__device__ __forceinline__ uint32_t pack2bf(float a, float b) {
    uint32_t ua = __builtin_bit_cast(uint32_t, a);
    uint32_t ub = __builtin_bit_cast(uint32_t, b);
#if defined(__has_builtin) && __has_builtin(__builtin_amdgcn_perm)
    // dst bytes = {S0.b3, S0.b2, S1.b3, S1.b2} with S0=ub, S1=ua
    return __builtin_amdgcn_perm(ub, ua, 0x07060302u);
#else
    return (ub & 0xFFFF0000u) | (ua >> 16);
#endif
}

struct Frag32B { uint4 lo, hi; };

// A fragment from LDS bf16: two contiguous 16B chunks -> 2x ds_load_b128.
__device__ __forceinline__ v16bf frag_from_lds(const unsigned short* p0,
                                               const unsigned short* p1) {
    Frag32B f;
    f.lo = *reinterpret_cast<const uint4*>(p0);
    f.hi = *reinterpret_cast<const uint4*>(p1);
    return __builtin_bit_cast(v16bf, f);
}

// B fragment from global fp32 weights: 4x global_load_b128 + 8x v_perm_b32.
__device__ __forceinline__ v16bf frag_from_f32(const float* __restrict__ p0,
                                               const float* __restrict__ p1) {
    float4 a0 = *reinterpret_cast<const float4*>(p0);
    float4 a1 = *reinterpret_cast<const float4*>(p0 + 4);
    float4 b0 = *reinterpret_cast<const float4*>(p1);
    float4 b1 = *reinterpret_cast<const float4*>(p1 + 4);
    v8u w;
    w[0] = pack2bf(a0.x, a0.y); w[1] = pack2bf(a0.z, a0.w);
    w[2] = pack2bf(a1.x, a1.y); w[3] = pack2bf(a1.z, a1.w);
    w[4] = pack2bf(b0.x, b0.y); w[5] = pack2bf(b0.z, b0.w);
    w[6] = pack2bf(b1.x, b1.y); w[7] = pack2bf(b1.z, b1.w);
    return __builtin_bit_cast(v16bf, w);
}

__device__ __forceinline__ v8f wmma_bf16(v16bf a, v16bf b, v8f c) {
    return __builtin_amdgcn_wmma_f32_16x16x32_bf16(false, a, false, b, (short)0, c,
                                                   false, false);
}

__device__ __forceinline__ float silu(float g) {
    return g / (1.0f + __expf(-g));
}

// CDNA5 async global->LDS b128 copy (TENSOR-free TDM path; tracked by ASYNCcnt).
__device__ __forceinline__ void async_g2l_b128(unsigned lds_byte_addr,
                                               const void* gaddr) {
    asm volatile("global_load_async_to_lds_b128 %0, %1, off"
                 :: "v"(lds_byte_addr), "v"(gaddr) : "memory");
}
__device__ __forceinline__ void wait_async0() {
    asm volatile("s_wait_asynccnt 0x0" ::: "memory");
}

// ---------------------------------------------------------------- x -> bf16
__global__ void k_convert_x(const float* __restrict__ x,
                            unsigned short* __restrict__ xb) {
    int i = (blockIdx.x * blockDim.x + threadIdx.x) * 4;
    float4 v = *reinterpret_cast<const float4*>(x + i);
    ushort4 o;
    o.x = f2bfbits(v.x); o.y = f2bfbits(v.y);
    o.z = f2bfbits(v.z); o.w = f2bfbits(v.w);
    *reinterpret_cast<ushort4*>(xb + i) = o;
}

__global__ void k_zero_counts(int* cnt) {
    if (threadIdx.x < NEXP) cnt[threadIdx.x] = 0;
}

// ---------------------------------------------------------------- router
__global__ __launch_bounds__(256) void k_router(const float* __restrict__ x,
                                                const float* __restrict__ rw,
                                                int* __restrict__ cnt,
                                                int* __restrict__ lists,
                                                float* __restrict__ gates) {
    const int wave = threadIdx.x >> 5;
    const int lane = threadIdx.x & 31;
    const int t = blockIdx.x * 8 + wave;
    const float* xt = x + (size_t)t * HDIM;

    float logit[NEXP];
#pragma unroll
    for (int e = 0; e < NEXP; ++e) {
        const float* we = rw + (size_t)e * HDIM;
        float s = 0.f;
        for (int h = lane; h < HDIM; h += 32) s += xt[h] * we[h];
#pragma unroll
        for (int off = 16; off; off >>= 1) s += __shfl_xor(s, off, 32);
        logit[e] = s;
    }
    int i0 = 0; float l0 = logit[0];
#pragma unroll
    for (int e = 1; e < NEXP; ++e) if (logit[e] > l0) { l0 = logit[e]; i0 = e; }
    int i1 = (i0 == 0) ? 1 : 0; float l1 = logit[i1];
#pragma unroll
    for (int e = 0; e < NEXP; ++e)
        if (e != i0 && logit[e] > l1) { l1 = logit[e]; i1 = e; }
    float e1 = __expf(l1 - l0);
    float w0 = 1.0f / (1.0f + e1);
    float w1 = e1 / (1.0f + e1);
    if (lane == 0) {
        int p0 = atomicAdd(&cnt[i0], 1);
        lists[i0 * T_TOK + p0] = t; gates[i0 * T_TOK + p0] = w0;
        int p1 = atomicAdd(&cnt[i1], 1);
        lists[i1 * T_TOK + p1] = t; gates[i1 * T_TOK + p1] = w1;
    }
}

// ---------------------------------------------------------------- shared expert
__global__ __launch_bounds__(256) void k_shared(const unsigned short* __restrict__ xb,
                                                const float* __restrict__ wgt,   // [ISH,H]
                                                const float* __restrict__ wup,   // [ISH,H]
                                                const float* __restrict__ wdn,   // [H,ISH]
                                                float* __restrict__ out) {
    __shared__ __align__(16) unsigned short xs[16 * HDIM];  // reused as h-tile
    const int t0 = blockIdx.x * 16;
    const int tid = threadIdx.x;
    const int wave = tid >> 5, lane = tid & 31;
    const int c = lane & 15, hi = lane >> 4;
    {   // async-stage the 32KB x-tile straight into LDS (no VGPR roundtrip)
        const char* gbase = (const char*)(xb + (size_t)t0 * HDIM);
        unsigned lbase = (unsigned)(uintptr_t)xs;
#pragma unroll
        for (int j = 0; j < 8; ++j) {
            int i = tid + j * 256;
            async_g2l_b128(lbase + i * 16, gbase + (size_t)i * 16);
        }
        wait_async0();
    }
    __syncthreads();

    const int n0 = wave * 128;
    v8f zf = {0.f, 0.f, 0.f, 0.f, 0.f, 0.f, 0.f, 0.f};
    v8f accg[8], accu[8];
#pragma unroll
    for (int i = 0; i < 8; ++i) { accg[i] = zf; accu[i] = zf; }

    for (int k0 = 0; k0 < HDIM; k0 += 32) {
        v16bf a = frag_from_lds(xs + c * HDIM + k0 + hi * 8,
                                xs + c * HDIM + k0 + 16 + hi * 8);
#pragma unroll
        for (int nt = 0; nt < 8; ++nt) {
            const int col = n0 + nt * 16 + c;
            const float* pg = wgt + (size_t)col * HDIM + k0;
            __builtin_prefetch(pg + 32, 0, 0);
            accg[nt] = wmma_bf16(a, frag_from_f32(pg + hi * 8, pg + 16 + hi * 8), accg[nt]);
            const float* pu = wup + (size_t)col * HDIM + k0;
            __builtin_prefetch(pu + 32, 0, 0);
            accu[nt] = wmma_bf16(a, frag_from_f32(pu + hi * 8, pu + 16 + hi * 8), accu[nt]);
        }
    }
    __syncthreads();  // done reading x-tile; reuse buffer for h-tile
    unsigned short* hs = xs;
#pragma unroll
    for (int nt = 0; nt < 8; ++nt)
#pragma unroll
        for (int v = 0; v < 8; ++v) {
            int m = v + 8 * hi;
            hs[m * ISH + n0 + nt * 16 + c] = f2bfbits(silu(accg[nt][v]) * accu[nt][v]);
        }
    __syncthreads();

    v8f accd[8];
#pragma unroll
    for (int i = 0; i < 8; ++i) accd[i] = zf;
    for (int k0 = 0; k0 < ISH; k0 += 32) {
        v16bf a = frag_from_lds(hs + c * ISH + k0 + hi * 8,
                                hs + c * ISH + k0 + 16 + hi * 8);
#pragma unroll
        for (int nt = 0; nt < 8; ++nt) {
            const int col = n0 + nt * 16 + c;
            const float* pd = wdn + (size_t)col * ISH + k0;
            __builtin_prefetch(pd + 32, 0, 0);
            accd[nt] = wmma_bf16(a, frag_from_f32(pd + hi * 8, pd + 16 + hi * 8), accd[nt]);
        }
    }
#pragma unroll
    for (int nt = 0; nt < 8; ++nt)
#pragma unroll
        for (int v = 0; v < 8; ++v) {
            int m = v + 8 * hi;
            out[(size_t)(t0 + m) * HDIM + n0 + nt * 16 + c] = accd[nt][v];
        }
}

// ---------------------------------------------------------------- routed experts
__global__ __launch_bounds__(256) void k_moe(const unsigned short* __restrict__ xb,
                                             const float* __restrict__ wg,  // [E,I,H]
                                             const float* __restrict__ wu,  // [E,I,H]
                                             const float* __restrict__ wd,  // [E,H,I]
                                             const int* __restrict__ cnt,
                                             const int* __restrict__ lists,
                                             const float* __restrict__ gates,
                                             float* __restrict__ out) {
    const int e = blockIdx.y;
    const int tile = blockIdx.x;
    const int count = cnt[e];
    if (tile * 16 >= count) return;

    __shared__ __align__(16) unsigned short xs[16 * HDIM];  // reused as h-tile
    __shared__ int toks[16];
    __shared__ float gws[16];
    const int tid = threadIdx.x;
    const int wave = tid >> 5, lane = tid & 31;
    const int c = lane & 15, hi = lane >> 4;

    if (tid < 16) {
        int s = tile * 16 + tid;
        if (s < count) { toks[tid] = lists[e * T_TOK + s]; gws[tid] = gates[e * T_TOK + s]; }
        else           { toks[tid] = lists[e * T_TOK + tile * 16]; gws[tid] = 0.f; }
    }
    __syncthreads();
    {   // async-gather 16 token rows into LDS
        int row = tid >> 4, part = tid & 15;
        const char* gsrc = (const char*)(xb + (size_t)toks[row] * HDIM);
        unsigned lrow = (unsigned)(uintptr_t)(xs + row * HDIM);
#pragma unroll
        for (int j = 0; j < 8; ++j) {
            int i = part + j * 16;
            async_g2l_b128(lrow + i * 16, gsrc + (size_t)i * 16);
        }
        wait_async0();
    }
    __syncthreads();

    const float* wge = wg + (size_t)e * IDIM * HDIM;
    const float* wue = wu + (size_t)e * IDIM * HDIM;
    const float* wde = wd + (size_t)e * HDIM * IDIM;

    v8f zf = {0.f, 0.f, 0.f, 0.f, 0.f, 0.f, 0.f, 0.f};
    const int n0 = wave * 64;
    v8f accg[4], accu[4];
#pragma unroll
    for (int i = 0; i < 4; ++i) { accg[i] = zf; accu[i] = zf; }

    for (int k0 = 0; k0 < HDIM; k0 += 32) {
        v16bf a = frag_from_lds(xs + c * HDIM + k0 + hi * 8,
                                xs + c * HDIM + k0 + 16 + hi * 8);
#pragma unroll
        for (int nt = 0; nt < 4; ++nt) {
            const int col = n0 + nt * 16 + c;
            const float* pg = wge + (size_t)col * HDIM + k0;
            __builtin_prefetch(pg + 32, 0, 0);
            accg[nt] = wmma_bf16(a, frag_from_f32(pg + hi * 8, pg + 16 + hi * 8), accg[nt]);
            const float* pu = wue + (size_t)col * HDIM + k0;
            __builtin_prefetch(pu + 32, 0, 0);
            accu[nt] = wmma_bf16(a, frag_from_f32(pu + hi * 8, pu + 16 + hi * 8), accu[nt]);
        }
    }
    __syncthreads();
    unsigned short* hs = xs;
#pragma unroll
    for (int nt = 0; nt < 4; ++nt)
#pragma unroll
        for (int v = 0; v < 8; ++v) {
            int m = v + 8 * hi;
            hs[m * IDIM + n0 + nt * 16 + c] = f2bfbits(silu(accg[nt][v]) * accu[nt][v]);
        }
    __syncthreads();

    const int h0 = wave * 128;
    v8f accd[8];
#pragma unroll
    for (int i = 0; i < 8; ++i) accd[i] = zf;
    for (int k0 = 0; k0 < IDIM; k0 += 32) {
        v16bf a = frag_from_lds(hs + c * IDIM + k0 + hi * 8,
                                hs + c * IDIM + k0 + 16 + hi * 8);
#pragma unroll
        for (int nt = 0; nt < 8; ++nt) {
            const int col = h0 + nt * 16 + c;
            const float* pd = wde + (size_t)col * IDIM + k0;
            __builtin_prefetch(pd + 32, 0, 0);
            accd[nt] = wmma_bf16(a, frag_from_f32(pd + hi * 8, pd + 16 + hi * 8), accd[nt]);
        }
    }
#pragma unroll
    for (int nt = 0; nt < 8; ++nt)
#pragma unroll
        for (int v = 0; v < 8; ++v) {
            int m = v + 8 * hi;
            atomicAdd(&out[(size_t)toks[m] * HDIM + h0 + nt * 16 + c],
                      gws[m] * accd[nt][v]);
        }
}

extern "C" void kernel_launch(void* const* d_in, const int* in_sizes, int n_in,
                              void* d_out, int out_size, void* d_ws, size_t ws_size,
                              hipStream_t stream) {
    (void)in_sizes; (void)n_in; (void)out_size; (void)ws_size;
    const float* x        = (const float*)d_in[0];
    const float* router_w = (const float*)d_in[1];
    const float* wg       = (const float*)d_in[2];
    const float* wu       = (const float*)d_in[3];
    const float* wd       = (const float*)d_in[4];
    const float* sw_gate  = (const float*)d_in[5];
    const float* sw_up    = (const float*)d_in[6];
    const float* sw_down  = (const float*)d_in[7];
    float* out = (float*)d_out;

    char* ws = (char*)d_ws;
    unsigned short* xb = (unsigned short*)ws;                         // T*H bf16 = 4 MB
    size_t off = (size_t)T_TOK * HDIM * sizeof(unsigned short);
    int* cnt = (int*)(ws + off);            off += 256;
    int* lists = (int*)(ws + off);          off += (size_t)NEXP * T_TOK * sizeof(int);
    float* gates = (float*)(ws + off);

    k_convert_x<<<(T_TOK * HDIM / 4) / 256, 256, 0, stream>>>(x, xb);
    k_zero_counts<<<1, 32, 0, stream>>>(cnt);
    k_router<<<T_TOK / 8, 256, 0, stream>>>(x, router_w, cnt, lists, gates);
    k_shared<<<T_TOK / 16, 256, 0, stream>>>(xb, sw_gate, sw_up, sw_down, out);
    k_moe<<<dim3(T_TOK / 16, NEXP), 256, 0, stream>>>(xb, wg, wu, wd, cnt, lists, gates, out);
}
